// TRAR_70652212019463
// MI455X (gfx1250) — compile-verified, matrix operand order checked by reading, not run on hardware
//
#include <hip/hip_runtime.h>

// ---------------------------------------------------------------------------
// TRAR block forward for MI455X (gfx1250, wave32, WMMA + TDM).
// Sizes: B=512, LX=64, LY=14, H=8, DH=64, D=512, FF=2048
// The FFT term is collapsed algebraically to real GEMMs:
//   sc = scale * [ (q @ rev(k)^T) @ C  +  (q @ k^T) * diag(w) ]
// with C a symmetric circulant: c_d = (-1)^d/64 * sum_j w_j cos(2*pi*d*j/64).
// ---------------------------------------------------------------------------

typedef __attribute__((ext_vector_type(16))) __bf16 v16bf;
typedef __attribute__((ext_vector_type(8)))  float  v8f;

#define DEVI __device__ __forceinline__

#if defined(__HIP_DEVICE_COMPILE__) && __has_builtin(__builtin_amdgcn_tensor_load_to_lds) && __has_builtin(__builtin_amdgcn_s_wait_tensorcnt)
#define USE_TDM 1
#else
#define USE_TDM 0
#endif

#if defined(__HIP_DEVICE_COMPILE__) && __has_builtin(__builtin_amdgcn_rcpf)
DEVI float fast_rcp(float x) { return __builtin_amdgcn_rcpf(x); }
#else
DEVI float fast_rcp(float x) { return 1.0f / x; }
#endif

#if USE_TDM
typedef __attribute__((ext_vector_type(4))) unsigned int u32x4;
typedef __attribute__((ext_vector_type(8))) int          i32x8;
typedef __attribute__((ext_vector_type(4))) int          i32x4;

// Issue a TDM load of a 2D bf16 tile [tile_y rows x tile_x cols] from global
// (row stride = stride_elems, in elements) into LDS at byte offset lds_off,
// stored densely row-major. D# layout per CDNA5 ISA 8.3/8.4 (2D: groups 2,3=0).
// This toolchain exposes the 6-arg builtin form (extra int32x8 group, zeroed).
DEVI void tdm_load_2d_bf16(unsigned lds_off, const void* gaddr,
                           unsigned tile_x, unsigned tile_y,
                           unsigned tensor_x, unsigned tensor_y,
                           unsigned long stride_elems) {
  unsigned long ga = (unsigned long)gaddr;
  u32x4 g0;
  g0[0] = 1u;                                  // count=1, user descriptor
  g0[1] = lds_off;                             // lds_addr (bytes)
  g0[2] = (unsigned)ga;                        // global_addr[31:0]
  g0[3] = (unsigned)(ga >> 32) | (2u << 30);   // global_addr[56:32] | type=2
  i32x8 g1;
  g1[0] = (int)(1u << 16);                     // data_size=1 (2 bytes); mask=0
  g1[1] = (int)((tensor_x & 0xFFFFu) << 16);   // tensor_dim0[15:0] in [31:16]
  g1[2] = (int)((tensor_x >> 16) | ((tensor_y & 0xFFFFu) << 16));
  g1[3] = (int)((tensor_y >> 16) | (tile_x << 16));   // tile_dim0
  g1[4] = (int)tile_y;                         // tile_dim1 (tile_dim2=0)
  g1[5] = (int)(unsigned)(stride_elems & 0xFFFFFFFFu);  // dim0_stride lo
  g1[6] = (int)((unsigned)(stride_elems >> 32) & 0xFFFFu);
  g1[7] = 0;
  i32x4 gz4 = {0, 0, 0, 0};
  i32x8 gz8 = {0, 0, 0, 0, 0, 0, 0, 0};
  __builtin_amdgcn_tensor_load_to_lds(g0, g1, gz4, gz4, gz8, 0);
}
#endif

// ---- WMMA wrapper: D = A(16x32 bf16) * B(32x16 bf16) + C(16x16 f32) -------
DEVI v8f wmma_bf16(v16bf a, v16bf b, v8f c) {
  return __builtin_amdgcn_wmma_f32_16x16x32_bf16(
      /*neg_a=*/false, a, /*neg_b=*/false, b,
      /*c_mod=*/(short)0, c, /*reuse_a=*/false, /*reuse_b=*/false);
}

// ---- Load a 16x32 bf16 fragment from an LDS tile stored row-major ---------
// Per ISA 16-bit A layout: lane L holds row (L&15); elems 0..7 -> k = half*8+0..7,
// elems 8..15 -> k = 16+half*8+0..7, half = L>>4. B uses same layout indexed by N.
DEVI v16bf frag_ld(const __bf16* tile, int row0, int stride, int kofs, int lane) {
  union { v16bf v; unsigned int u[8]; } f;
  const __bf16* p = tile + (row0 + (lane & 15)) * stride + kofs;
  const int half = (lane >> 4) << 3;
#pragma unroll
  for (int j = 0; j < 4; ++j) {
    f.u[j]     = *reinterpret_cast<const unsigned int*>(p + half + 2 * j);
    f.u[4 + j] = *reinterpret_cast<const unsigned int*>(p + 16 + half + 2 * j);
  }
  return f.v;
}

// ---------------------------------------------------------------------------
// Elementwise f32 -> bf16 cast
// ---------------------------------------------------------------------------
__global__ void k_cast_bf16(const float* __restrict__ s, __bf16* __restrict__ d,
                            long n) {
  long i = (long)blockIdx.x * blockDim.x + threadIdx.x;
  long str = (long)gridDim.x * blockDim.x;
  for (; i < n; i += str) d[i] = (__bf16)s[i];
}

// ---------------------------------------------------------------------------
// Generic GEMM: C[M,N] = act(A[M,K] @ Bw[K,N] + bias[N])
// A,Bw bf16 row-major. Outputs: Cf (f32) and/or Cb (bf16), either may be null.
// Block tile 64x64, 4 waves, each wave a 32x32 quadrant (2x2 WMMA tiles).
// A tile staged by TDM (tensor_load_to_lds) when available.
// act: 0 = none, 1 = relu, 2 = sigmoid
// ---------------------------------------------------------------------------
__global__ __launch_bounds__(128) void k_gemm(
    const __bf16* __restrict__ A, const __bf16* __restrict__ Bw,
    const float* __restrict__ bias,
    float* __restrict__ Cf, __bf16* __restrict__ Cb,
    int M, int N, int K, int act) {
  __shared__ __bf16 As[64][32];
  __shared__ __bf16 Bts[64][32];  // transposed: [n][k]
  const int n0 = blockIdx.x * 64;
  const int m0 = blockIdx.y * 64;
  const int t = threadIdx.x;
  const int lane = t & 31;
  const int wid = t >> 5;
  const int wm = (wid >> 1) * 32;
  const int wn = (wid & 1) * 32;
  v8f acc[2][2] = {};

  for (int k0 = 0; k0 < K; k0 += 32) {
    if (k0 + 32 < K)  // gfx1250 global_prefetch_b8 for next K tile
      __builtin_prefetch((const void*)(A + (long)m0 * K + k0 + 32), 0, 1);
#if USE_TDM
    // ---- stage A tile 64x32 via Tensor Data Mover (wave 0 issues) ----
    if (wid == 0) {
      tdm_load_2d_bf16((unsigned)(size_t)&As[0][0],
                       (const void*)(A + (long)m0 * K + k0),
                       /*tile_x=*/32, /*tile_y=*/64,
                       /*tensor_x=*/(unsigned)K, /*tensor_y=*/(unsigned)M,
                       /*stride=*/(unsigned long)K);
    }
#else
    // ---- stage A tile 64x32 (2 uint4 chunks per thread) ----
#pragma unroll
    for (int i = 0; i < 2; ++i) {
      int cid = t + 128 * i;               // 0..255 chunks of 8 bf16
      int r = cid >> 2, c = (cid & 3) << 3;
      uint4 val = *reinterpret_cast<const uint4*>(A + (long)(m0 + r) * K + k0 + c);
      *reinterpret_cast<uint4*>(&As[r][c]) = val;
    }
#endif
    // ---- stage B tile 32x64 transposed into Bts[64][32] ----
    {
      int r = t >> 2, c = (t & 3) << 4;    // r in 0..31, c in {0,16,32,48}
      union { uint4 q[2]; __bf16 h[16]; } tmp;
      const __bf16* gp = Bw + (long)(k0 + r) * N + n0 + c;
      tmp.q[0] = *reinterpret_cast<const uint4*>(gp);
      tmp.q[1] = *reinterpret_cast<const uint4*>(gp + 8);
#pragma unroll
      for (int e = 0; e < 16; ++e) Bts[c + e][r] = tmp.h[e];
    }
#if USE_TDM
    if (wid == 0) __builtin_amdgcn_s_wait_tensorcnt(0);
#endif
    __syncthreads();
    v16bf af[2], bf[2];
#pragma unroll
    for (int mt = 0; mt < 2; ++mt)
      af[mt] = frag_ld(&As[0][0], wm + mt * 16, 32, 0, lane);
#pragma unroll
    for (int nt = 0; nt < 2; ++nt)
      bf[nt] = frag_ld(&Bts[0][0], wn + nt * 16, 32, 0, lane);
#pragma unroll
    for (int mt = 0; mt < 2; ++mt)
#pragma unroll
      for (int nt = 0; nt < 2; ++nt)
        acc[mt][nt] = wmma_bf16(af[mt], bf[nt], acc[mt][nt]);
    __syncthreads();
  }

  // ---- epilogue: C/D f32 layout: lane L -> n = L&15; vgpr r -> m = r + 8*(L>>4)
#pragma unroll
  for (int mt = 0; mt < 2; ++mt)
#pragma unroll
    for (int nt = 0; nt < 2; ++nt) {
      const int n = wn + nt * 16 + (lane & 15);
      const int mb = wm + mt * 16 + ((lane >> 4) << 3);
      const float bval = bias ? bias[n0 + n] : 0.0f;
#pragma unroll
      for (int r = 0; r < 8; ++r) {
        float v = acc[mt][nt][r] + bval;
        if (act == 1) v = v > 0.0f ? v : 0.0f;
        else if (act == 2) v = fast_rcp(1.0f + __expf(-v));
        const long idx = (long)(m0 + mb + r) * N + n0 + n;
        if (Cf) Cf[idx] = v;
        if (Cb) Cb[idx] = (__bf16)v;
      }
    }
}

// ---------------------------------------------------------------------------
// Pool: S[B,14,512] f32 -> mean over LY -> P[B*8,64] bf16 (flat identical idx)
// ---------------------------------------------------------------------------
__global__ void k_pool(const float* __restrict__ S, __bf16* __restrict__ P) {
  int i = blockIdx.x * blockDim.x + threadIdx.x;
  if (i >= 512 * 512) return;
  int b = i >> 9, c = i & 511;
  float s = 0.0f;
  for (int l = 0; l < 14; ++l) s += S[((long)b * 14 + l) * 512 + c];
  P[i] = (__bf16)(s * (1.0f / 14.0f));
}

// ---------------------------------------------------------------------------
// Fourier self-attention, one block per (b,h). 128 threads = 4 waves.
// q,k tiles staged via TDM; v staged transposed by VALU.
// ---------------------------------------------------------------------------
__global__ __launch_bounds__(128) void k_fattn(
    const __bf16* __restrict__ Q, const __bf16* __restrict__ Kq,
    const __bf16* __restrict__ V, const float* __restrict__ Wg,
    const unsigned char* __restrict__ xm, __bf16* __restrict__ Out) {
  __shared__ __align__(16) unsigned char sm[49152 + 512];
  __bf16* qs   = (__bf16*)(sm + 0);       // [64][64] 8KB
  __bf16* qrev = (__bf16*)(sm + 8192);    // [64][64] 8KB
  __bf16* ks   = (__bf16*)(sm + 16384);   // [64][64] 8KB
  __bf16* vt   = (__bf16*)(sm + 24576);   // [dh][l]  8KB
  __bf16* Cs   = (__bf16*)(sm + 32768);   // [64][64] 8KB circulant
  __bf16* a1s  = (__bf16*)(sm + 40960);   // [64][64] 8KB
  float*  scf  = (float*) (sm + 0);       // overlay qs+qrev (16KB), used late
  __bf16* ps   = (__bf16*)(sm + 16384);   // overlay ks (8KB), used late
  float*  c_sh = (float*) (sm + 49152);   // [64]
  float*  w_sh = (float*) (sm + 49408);   // [64]

  const int bh = blockIdx.x, b = bh >> 3, h = bh & 7;
  const int t = threadIdx.x, lane = t & 31, wid = t >> 5;
  const int wm = (wid >> 1) * 32, wn = (wid & 1) * 32;
  const long rowQ = (long)b * 64;

  if (t < 64) w_sh[t] = Wg[(long)bh * 64 + t];
#if USE_TDM
  if (wid == 0) {
    tdm_load_2d_bf16((unsigned)(size_t)qs, (const void*)(Q + rowQ * 512 + h * 64),
                     64, 64, 512, 64, 512);
    tdm_load_2d_bf16((unsigned)(size_t)ks, (const void*)(Kq + rowQ * 512 + h * 64),
                     64, 64, 512, 64, 512);
  }
#endif
  // ---- stage v (transposed); q,k direct if no TDM ----
#pragma unroll
  for (int i = 0; i < 4; ++i) {
    int cid = t + 128 * i;             // 0..511 chunks of 8 bf16
    int r = cid >> 3, c = (cid & 7) << 3;
    long g = (rowQ + r) * 512 + h * 64 + c;
#if !USE_TDM
    *reinterpret_cast<uint4*>(&qs[r * 64 + c]) = *reinterpret_cast<const uint4*>(Q + g);
    *reinterpret_cast<uint4*>(&ks[r * 64 + c]) = *reinterpret_cast<const uint4*>(Kq + g);
#endif
    union { uint4 q; __bf16 hh[8]; } tv;
    tv.q = *reinterpret_cast<const uint4*>(V + g);
#pragma unroll
    for (int e = 0; e < 8; ++e) vt[(c + e) * 64 + r] = tv.hh[e];
  }
#if USE_TDM
  if (wid == 0) __builtin_amdgcn_s_wait_tensorcnt(0);
#endif
  __syncthreads();
  // ---- build q_rev (column reversal mod 64) and circulant weights ----
  for (int i = 0; i < 32; ++i) {
    int idx = t * 32 + i, r = idx >> 6, n = idx & 63;
    qrev[r * 64 + n] = qs[r * 64 + ((64 - n) & 63)];
  }
  if (t < 64) {
    float s = 0.0f;
    for (int j = 0; j < 64; ++j)
      s += w_sh[j] * __cosf(0.09817477042f * (float)((t * j) & 63));
    c_sh[t] = ((t & 1) ? -1.0f : 1.0f) * (1.0f / 64.0f) * s;
  }
  __syncthreads();
  for (int i = 0; i < 32; ++i) {
    int idx = t * 32 + i, r = idx >> 6, n = idx & 63;
    Cs[r * 64 + n] = (__bf16)c_sh[(r - n) & 63];
  }
  __syncthreads();

  // ---- A1 = qrev @ k^T, A2 = q @ k^T (both 64x64, K=64) ----
  v8f a1[2][2] = {}, a2[2][2] = {};
#pragma unroll
  for (int kt = 0; kt < 2; ++kt) {
    v16bf fq[2], fr[2], fk[2];
#pragma unroll
    for (int mt = 0; mt < 2; ++mt) {
      fq[mt] = frag_ld(qs,   wm + mt * 16, 64, kt * 32, lane);
      fr[mt] = frag_ld(qrev, wm + mt * 16, 64, kt * 32, lane);
    }
#pragma unroll
    for (int nt = 0; nt < 2; ++nt)
      fk[nt] = frag_ld(ks, wn + nt * 16, 64, kt * 32, lane);
#pragma unroll
    for (int mt = 0; mt < 2; ++mt)
#pragma unroll
      for (int nt = 0; nt < 2; ++nt) {
        a2[mt][nt] = wmma_bf16(fq[mt], fk[nt], a2[mt][nt]);
        a1[mt][nt] = wmma_bf16(fr[mt], fk[nt], a1[mt][nt]);
      }
  }
  // ---- spill A1 to LDS (bf16) for the circulant GEMM ----
#pragma unroll
  for (int mt = 0; mt < 2; ++mt)
#pragma unroll
    for (int nt = 0; nt < 2; ++nt) {
      int n = wn + nt * 16 + (lane & 15);
      int mb = wm + mt * 16 + ((lane >> 4) << 3);
#pragma unroll
      for (int r = 0; r < 8; ++r) a1s[(mb + r) * 64 + n] = (__bf16)a1[mt][nt][r];
    }
  __syncthreads();
  // ---- S = A1 @ C (C symmetric, so row-major Cs doubles as Bt) ----
  v8f aS[2][2] = {};
#pragma unroll
  for (int kt = 0; kt < 2; ++kt) {
    v16bf fa[2], fc[2];
#pragma unroll
    for (int mt = 0; mt < 2; ++mt) fa[mt] = frag_ld(a1s, wm + mt * 16, 64, kt * 32, lane);
#pragma unroll
    for (int nt = 0; nt < 2; ++nt) fc[nt] = frag_ld(Cs,  wn + nt * 16, 64, kt * 32, lane);
#pragma unroll
    for (int mt = 0; mt < 2; ++mt)
#pragma unroll
      for (int nt = 0; nt < 2; ++nt)
        aS[mt][nt] = wmma_bf16(fa[mt], fc[nt], aS[mt][nt]);
  }
  // ---- sc = scale*(S + A2*w[n]) with mask; write f32 (overlays qs/qrev) ----
#pragma unroll
  for (int mt = 0; mt < 2; ++mt)
#pragma unroll
    for (int nt = 0; nt < 2; ++nt) {
      int n = wn + nt * 16 + (lane & 15);
      int mb = wm + mt * 16 + ((lane >> 4) << 3);
      float wn_ = w_sh[n];
      bool masked = xm[(long)b * 64 + n] != 0;
#pragma unroll
      for (int r = 0; r < 8; ++r) {
        float v = 0.125f * (aS[mt][nt][r] + a2[mt][nt][r] * wn_);
        scf[(mb + r) * 64 + n] = masked ? -1e9f : v;
      }
    }
  __syncthreads();
  // ---- softmax over key axis; write P bf16 (overlays ks) ----
  if (t < 64) {
    float mx = -3.4e38f;
    for (int j = 0; j < 64; ++j) mx = fmaxf(mx, scf[t * 64 + j]);
    float sum = 0.0f;
    for (int j = 0; j < 64; ++j) sum += __expf(scf[t * 64 + j] - mx);
    float inv = fast_rcp(sum);
    for (int j = 0; j < 64; ++j)
      ps[t * 64 + j] = (__bf16)(__expf(scf[t * 64 + j] - mx) * inv);
  }
  __syncthreads();
  // ---- O = P @ V ----
  v8f aO[2][2] = {};
#pragma unroll
  for (int kt = 0; kt < 2; ++kt) {
    v16bf fp[2], fv[2];
#pragma unroll
    for (int mt = 0; mt < 2; ++mt) fp[mt] = frag_ld(ps, wm + mt * 16, 64, kt * 32, lane);
#pragma unroll
    for (int nt = 0; nt < 2; ++nt) fv[nt] = frag_ld(vt, wn + nt * 16, 64, kt * 32, lane);
#pragma unroll
    for (int mt = 0; mt < 2; ++mt)
#pragma unroll
      for (int nt = 0; nt < 2; ++nt)
        aO[mt][nt] = wmma_bf16(fp[mt], fv[nt], aO[mt][nt]);
  }
#pragma unroll
  for (int mt = 0; mt < 2; ++mt)
#pragma unroll
    for (int nt = 0; nt < 2; ++nt) {
      int n = wn + nt * 16 + (lane & 15);
      int mb = wm + mt * 16 + ((lane >> 4) << 3);
#pragma unroll
      for (int r = 0; r < 8; ++r)
        Out[(rowQ + mb + r) * 512 + h * 64 + n] = (__bf16)aO[mt][nt][r];
    }
}

// ---------------------------------------------------------------------------
// Cross attention, one block per (b,h). LY=14 padded to 16 (scores) / 32 (K).
// ---------------------------------------------------------------------------
__global__ __launch_bounds__(128) void k_xattn(
    const __bf16* __restrict__ Q2, const __bf16* __restrict__ K2,
    const __bf16* __restrict__ V2, const unsigned char* __restrict__ ym,
    __bf16* __restrict__ Out) {
  __shared__ __bf16 q2s[64][64];
  __shared__ __bf16 k2s[16][64];
  __shared__ __bf16 v2t[64][32];   // [dh][j], j padded to 32
  __shared__ float  sc2[64][16];
  __shared__ __bf16 ps2[64][32];   // [m][j], j padded to 32
  const int bh = blockIdx.x, b = bh >> 3, h = bh & 7;
  const int t = threadIdx.x, lane = t & 31, wid = t >> 5;

#if USE_TDM
  if (wid == 0) {
    tdm_load_2d_bf16((unsigned)(size_t)&q2s[0][0],
                     (const void*)(Q2 + ((long)b * 64) * 512 + h * 64),
                     64, 64, 512, 64, 512);
  }
#endif
  // zero padded tiles
  for (int i = 0; i < 16; ++i) {
    (&v2t[0][0])[t + 128 * i] = (__bf16)0.0f;
    (&ps2[0][0])[t + 128 * i] = (__bf16)0.0f;
  }
#if !USE_TDM
  // q2 tile
#pragma unroll
  for (int i = 0; i < 4; ++i) {
    int cid = t + 128 * i, r = cid >> 3, c = (cid & 7) << 3;
    *reinterpret_cast<uint4*>(&q2s[r][c]) =
        *reinterpret_cast<const uint4*>(Q2 + ((long)b * 64 + r) * 512 + h * 64 + c);
  }
#endif
  // k2 tile [16][64], rows >=14 zero
  {
    int r = t >> 3, c = (t & 7) << 3;
    uint4 val = make_uint4(0, 0, 0, 0);
    if (r < 14)
      val = *reinterpret_cast<const uint4*>(K2 + ((long)b * 14 + r) * 512 + h * 64 + c);
    *reinterpret_cast<uint4*>(&k2s[r][c]) = val;
  }
  __syncthreads();  // v2t zeros visible before scatter fill
  // v2 transposed fill (896 elems)
  for (int i = 0; i < 7; ++i) {
    int idx = t + 128 * i, j = idx >> 6, dh = idx & 63;
    v2t[dh][j] = V2[((long)b * 14 + j) * 512 + h * 64 + dh];
  }
#if USE_TDM
  if (wid == 0) __builtin_amdgcn_s_wait_tensorcnt(0);
#endif
  __syncthreads();

  // ---- sc2 = scale * q2 @ k2^T : wave wid owns m-tile wid, single n-tile ----
  v8f s1 = {};
#pragma unroll
  for (int kt = 0; kt < 2; ++kt) {
    v16bf fa = frag_ld(&q2s[0][0], wid * 16, 64, kt * 32, lane);
    v16bf fb = frag_ld(&k2s[0][0], 0, 64, kt * 32, lane);
    s1 = wmma_bf16(fa, fb, s1);
  }
  {
    int j = lane & 15;
    int mb = wid * 16 + ((lane >> 4) << 3);
    bool masked = (j < 14) && (ym[(long)b * 14 + j] != 0);
#pragma unroll
    for (int r = 0; r < 8; ++r) {
      float v = 0.125f * s1[r];
      sc2[mb + r][j] = masked ? -1e9f : v;
    }
  }
  __syncthreads();
  if (t < 64) {
    float mx = -3.4e38f;
    for (int j = 0; j < 14; ++j) mx = fmaxf(mx, sc2[t][j]);
    float sum = 0.0f;
    for (int j = 0; j < 14; ++j) sum += __expf(sc2[t][j] - mx);
    float inv = fast_rcp(sum);
    for (int j = 0; j < 14; ++j)
      ps2[t][j] = (__bf16)(__expf(sc2[t][j] - mx) * inv);
  }
  __syncthreads();
  // ---- O = P @ v2 : wave wid owns m-tile wid, 4 n-tiles, K=32 (padded) ----
  v8f o[4] = {};
  v16bf fa = frag_ld(&ps2[0][0], wid * 16, 32, 0, lane);
#pragma unroll
  for (int nt = 0; nt < 4; ++nt) {
    v16bf fb = frag_ld(&v2t[0][0], nt * 16, 32, 0, lane);
    o[nt] = wmma_bf16(fa, fb, o[nt]);
  }
  {
    int mb = wid * 16 + ((lane >> 4) << 3);
#pragma unroll
    for (int nt = 0; nt < 4; ++nt) {
      int n = nt * 16 + (lane & 15);
#pragma unroll
      for (int r = 0; r < 8; ++r)
        Out[((long)b * 64 + mb + r) * 512 + h * 64 + n] = (__bf16)o[nt][r];
    }
  }
}

// ---------------------------------------------------------------------------
// Residual + LayerNorm (torch-style: std with ddof=1, eps added to std).
// One block per row of 512. Optional f32 and bf16 outputs.
// ---------------------------------------------------------------------------
__global__ __launch_bounds__(128) void k_resln(
    const float* __restrict__ base, const float* __restrict__ delta,
    const float* __restrict__ g, const float* __restrict__ bt,
    float* __restrict__ of, __bf16* __restrict__ ob) {
  __shared__ float red[128];
  const long row = blockIdx.x;
  const int t = threadIdx.x;
  float v[4];
  float s = 0.0f;
#pragma unroll
  for (int i = 0; i < 4; ++i) {
    int c = t + 128 * i;
    v[i] = base[row * 512 + c] + delta[row * 512 + c];
    s += v[i];
  }
  red[t] = s;
  __syncthreads();
  for (int off = 64; off > 0; off >>= 1) {
    if (t < off) red[t] += red[t + off];
    __syncthreads();
  }
  const float mean = red[0] * (1.0f / 512.0f);
  __syncthreads();
  s = 0.0f;
#pragma unroll
  for (int i = 0; i < 4; ++i) {
    float d = v[i] - mean;
    s += d * d;
  }
  red[t] = s;
  __syncthreads();
  for (int off = 64; off > 0; off >>= 1) {
    if (t < off) red[t] += red[t + off];
    __syncthreads();
  }
  const float sd = sqrtf(red[0] * (1.0f / 511.0f));
  const float inv = 1.0f / (sd + 1e-6f);
#pragma unroll
  for (int i = 0; i < 4; ++i) {
    int c = t + 128 * i;
    float o = g[c] * (v[i] - mean) * inv + bt[c];
    if (of) of[row * 512 + c] = o;
    if (ob) ob[row * 512 + c] = (__bf16)o;
  }
}

// ---------------------------------------------------------------------------
// Host orchestration
// ---------------------------------------------------------------------------
extern "C" void kernel_launch(void* const* d_in, const int* in_sizes, int n_in,
                              void* d_out, int out_size, void* d_ws, size_t ws_size,
                              hipStream_t stream) {
  (void)in_sizes; (void)n_in; (void)out_size; (void)ws_size;
  const int MX = 512 * 64;   // 32768
  const int MY = 512 * 14;   // 7168
  const int BH = 512 * 8;    // 4096

  const float* x    = (const float*)d_in[0];
  const float* y    = (const float*)d_in[1];
  const unsigned char* xm = (const unsigned char*)d_in[2];
  const unsigned char* ym = (const unsigned char*)d_in[3];
  const float* f_wq = (const float*)d_in[4];  const float* f_bq = (const float*)d_in[5];
  const float* f_wk = (const float*)d_in[6];  const float* f_bk = (const float*)d_in[7];
  const float* f_wv = (const float*)d_in[8];  const float* f_bv = (const float*)d_in[9];
  const float* f_wm = (const float*)d_in[10]; const float* f_bm = (const float*)d_in[11];
  const float* f_ws = (const float*)d_in[12]; const float* f_bs = (const float*)d_in[13];
  const float* f_wws= (const float*)d_in[14]; const float* f_bws= (const float*)d_in[15];
  const float* m_wq = (const float*)d_in[16]; const float* m_bq = (const float*)d_in[17];
  const float* m_wk = (const float*)d_in[18]; const float* m_bk = (const float*)d_in[19];
  const float* m_wv = (const float*)d_in[20]; const float* m_bv = (const float*)d_in[21];
  const float* m_wm = (const float*)d_in[22]; const float* m_bm = (const float*)d_in[23];
  const float* w1   = (const float*)d_in[24]; const float* b1   = (const float*)d_in[25];
  const float* w2   = (const float*)d_in[26]; const float* b2   = (const float*)d_in[27];
  const float* ln1g = (const float*)d_in[28]; const float* ln1b = (const float*)d_in[29];
  const float* ln2g = (const float*)d_in[30]; const float* ln2b = (const float*)d_in[31];
  const float* ln3g = (const float*)d_in[32]; const float* ln3b = (const float*)d_in[33];

  char* ws = (char*)d_ws;
  size_t off = 0;
  auto alloc = [&](size_t bytes) -> char* {
    char* p = ws + off;
    off = (off + bytes + 255) & ~(size_t)255;
    return p;
  };
  // bf16 weights
  __bf16* wqb = (__bf16*)alloc(512 * 512 * 2);
  __bf16* wkb = (__bf16*)alloc(512 * 512 * 2);
  __bf16* wvb = (__bf16*)alloc(512 * 512 * 2);
  __bf16* wmb = (__bf16*)alloc(512 * 512 * 2);
  __bf16* wsb = (__bf16*)alloc(512 * 512 * 2);
  __bf16* wwsb= (__bf16*)alloc(64 * 64 * 2);
  __bf16* mqb = (__bf16*)alloc(512 * 512 * 2);
  __bf16* mkb = (__bf16*)alloc(512 * 512 * 2);
  __bf16* mvb = (__bf16*)alloc(512 * 512 * 2);
  __bf16* mmb = (__bf16*)alloc(512 * 512 * 2);
  __bf16* w1b = (__bf16*)alloc(512 * 2048 * 2);
  __bf16* w2b = (__bf16*)alloc(2048 * 512 * 2);
  // bf16 activations
  __bf16* xb  = (__bf16*)alloc((size_t)MX * 512 * 2);
  __bf16* yb  = (__bf16*)alloc((size_t)MY * 512 * 2);
  // reusable region: Qb..Att1b spans >= FF hidden (134MB)
  __bf16* Qb   = (__bf16*)alloc((size_t)MX * 512 * 2);
  __bf16* Kb   = (__bf16*)alloc((size_t)MX * 512 * 2);
  __bf16* Vb   = (__bf16*)alloc((size_t)MX * 512 * 2);
  float*  Sf   = (float*) alloc((size_t)MY * 512 * 4);   // reused as K2b/V2b
  __bf16* poolb= (__bf16*)alloc((size_t)BH * 64 * 2);
  float*  Wgtf = (float*) alloc((size_t)BH * 64 * 4);
  __bf16* Att1b= (__bf16*)alloc((size_t)MX * 512 * 2);   // reused as Att2b
  // big f32 buffers
  float*  Tmpf = (float*) alloc((size_t)MX * 512 * 4);
  float*  X1f  = (float*) alloc((size_t)MX * 512 * 4);
  __bf16* X1b  = (__bf16*)alloc((size_t)MX * 512 * 2);
  float*  X2f  = (float*) alloc((size_t)MX * 512 * 4);
  __bf16* X2b  = (__bf16*)alloc((size_t)MX * 512 * 2);
  // aliases (sequenced reuse)
  __bf16* K2b  = (__bf16*)Sf;
  __bf16* V2b  = (__bf16*)((char*)Sf + (size_t)MY * 512 * 2);
  __bf16* Q2b  = Qb;
  __bf16* Att2b= Att1b;
  __bf16* Hb   = Qb;   // [MX, 2048] bf16 overlays Qb..Att1b span

  auto cast = [&](const float* s, __bf16* d, long n) {
    k_cast_bf16<<<2048, 256, 0, stream>>>(s, d, n);
  };
  auto gemm = [&](const __bf16* A, const __bf16* W, const float* bias,
                  float* Cf, __bf16* Cb, int M, int N, int K, int act) {
    dim3 g(N / 64, M / 64);
    k_gemm<<<g, 128, 0, stream>>>(A, W, bias, Cf, Cb, M, N, K, act);
  };

  // ---- casts ----
  cast(x, xb, (long)MX * 512);
  cast(y, yb, (long)MY * 512);
  cast(f_wq, wqb, 512 * 512); cast(f_wk, wkb, 512 * 512);
  cast(f_wv, wvb, 512 * 512); cast(f_wm, wmb, 512 * 512);
  cast(f_ws, wsb, 512 * 512); cast(f_wws, wwsb, 64 * 64);
  cast(m_wq, mqb, 512 * 512); cast(m_wk, mkb, 512 * 512);
  cast(m_wv, mvb, 512 * 512); cast(m_wm, mmb, 512 * 512);
  cast(w1, w1b, 512 * 2048);  cast(w2, w2b, 2048 * 512);

  // ---- Fourier self-attention branch ----
  gemm(xb, wqb, f_bq, nullptr, Qb, MX, 512, 512, 0);
  gemm(xb, wkb, f_bk, nullptr, Kb, MX, 512, 512, 0);
  gemm(xb, wvb, f_bv, nullptr, Vb, MX, 512, 512, 0);
  gemm(yb, wsb, f_bs, Sf, nullptr, MY, 512, 512, 0);
  k_pool<<<(512 * 512) / 256, 256, 0, stream>>>(Sf, poolb);
  gemm(poolb, wwsb, f_bws, Wgtf, nullptr, BH, 64, 64, 2 /*sigmoid*/);
  k_fattn<<<BH, 128, 0, stream>>>(Qb, Kb, Vb, Wgtf, xm, Att1b);
  gemm(Att1b, wmb, f_bm, Tmpf, nullptr, MX, 512, 512, 0);
  k_resln<<<MX, 128, 0, stream>>>(x, Tmpf, ln1g, ln1b, X1f, X1b);

  // ---- cross attention ----
  gemm(X1b, mqb, m_bq, nullptr, Q2b, MX, 512, 512, 0);
  gemm(yb, mkb, m_bk, nullptr, K2b, MY, 512, 512, 0);
  gemm(yb, mvb, m_bv, nullptr, V2b, MY, 512, 512, 0);
  k_xattn<<<BH, 128, 0, stream>>>(Q2b, K2b, V2b, ym, Att2b);
  gemm(Att2b, mmb, m_bm, Tmpf, nullptr, MX, 512, 512, 0);
  k_resln<<<MX, 128, 0, stream>>>(X1f, Tmpf, ln2g, ln2b, X2f, X2b);

  // ---- FFN ----
  gemm(X2b, w1b, b1, nullptr, Hb, MX, 2048, 512, 1 /*relu*/);
  gemm(Hb, w2b, b2, Tmpf, nullptr, MX, 512, 2048, 0);
  k_resln<<<MX, 128, 0, stream>>>(X2f, Tmpf, ln3g, ln3b, (float*)d_out, nullptr);
}
